// DoubleConv_24713241821821
// MI455X (gfx1250) — matrix operand order repaired and checked by reading, not compiled, not run on hardware
//
#include <hip/hip_runtime.h>
#include <hip/hip_bf16.h>
#include <math.h>

// ---------------------------------------------------------------------------
// Types for CDNA5 WMMA (wave32)
// ---------------------------------------------------------------------------
typedef __attribute__((ext_vector_type(16))) __bf16 v16bf;
typedef __attribute__((ext_vector_type(8)))  float  v8f;

// f32x2 -> packed bf16x2: round-half-up then byte-select via v_perm_b32.
// 3 VALU ops total (vs ~10 for the tie-to-even bit sequence).
__device__ __forceinline__ unsigned int pack_bf16x2(float a, float b) {
    unsigned int ua = __float_as_uint(a) + 0x8000u;
    unsigned int ub = __float_as_uint(b) + 0x8000u;
    // result bytes: {ub[3], ub[2], ua[3], ua[2]}  (src0=ub -> bytes 4..7)
    return __builtin_amdgcn_perm(ub, ua, 0x07060302u);
}

union F4 { float4 v; float f[4]; };

// ---------------------------------------------------------------------------
// Convert fp32 weights -> bf16 (row-major [N,K] preserved).
// ---------------------------------------------------------------------------
__global__ void f32_to_bf16_kernel(const float* __restrict__ src,
                                   unsigned short* __restrict__ dst, int n) {
    int i = blockIdx.x * blockDim.x + threadIdx.x;
    if (i < n)
        dst[i] = (unsigned short)((__float_as_uint(src[i]) + 0x8000u) >> 16);
}

// ---------------------------------------------------------------------------
// GEMM: out[M,N] = A[M,K] @ Wb[N,K]^T (+bias).  A fp32, row stride == K.
// Block = 128 threads (4 waves) -> 64x64 output tile:
//   wave w owns columns n_base..n_base+15 and FOUR 16-row M-subtiles,
//   reusing one B fragment across 4 WMMAs per K-step.
// A tile (64 rows x 32 k, bf16) staged in LDS once per K-step.
// ---------------------------------------------------------------------------
template <int K, int LDO, bool BIAS>
__global__ __launch_bounds__(128)
void gemm_wmma_bf16(const float* __restrict__ A,
                    const unsigned short* __restrict__ Wb,
                    const float* __restrict__ bias,
                    float* __restrict__ out, int N) {
    __shared__ unsigned int lds_a[64 * 16];   // 64 rows x 32 bf16 (4 KB)

    const int m_base = blockIdx.x * 64;
    const int wv     = threadIdx.x >> 5;
    const int lane   = threadIdx.x & 31;
    const int n_base = blockIdx.y * 64 + wv * 16;
    const int col    = n_base + (lane & 15);
    const bool colok = (col < N);
    const int colc   = colok ? col : (N - 1);   // clamped address; result unused

    // A staging: thread t covers row (t>>1), k span (t&1)*16 .. +15
    const int tr = threadIdx.x >> 1;
    const int tk = (threadIdx.x & 1) * 16;

    // A fragment base within a 16-row subtile (16-bit A 16x32 layout):
    //  lane<16: row=lane,  dwords {0..3}->K0..7, {8..11}->K16..23
    //  lane>=16: row=lane-16, K8..15 / K24..31 (base +4)
    const int abase0 = (lane & 15) * 16 + ((lane < 16) ? 0 : 4);

    const uint4* __restrict__ wrow = (const uint4*)(Wb + (size_t)colc * K);

    v8f acc[4];
#pragma unroll
    for (int s = 0; s < 4; ++s) acc[s] = (v8f){0.f,0.f,0.f,0.f,0.f,0.f,0.f,0.f};
    union { v16bf v; uint4 q[2]; } afrag, bfrag;

#pragma unroll
    for (int k0 = 0; k0 < K; k0 += 32) {
        // ---- stage 64x32 A tile (f32 -> bf16) into LDS ----
        {
            const float4* ap = (const float4*)(A + (size_t)(m_base + tr) * K + k0 + tk);
            const float4 a0 = ap[0], a1 = ap[1], a2 = ap[2], a3 = ap[3];
            uint4 p0, p1;
            p0.x = pack_bf16x2(a0.x, a0.y); p0.y = pack_bf16x2(a0.z, a0.w);
            p0.z = pack_bf16x2(a1.x, a1.y); p0.w = pack_bf16x2(a1.z, a1.w);
            p1.x = pack_bf16x2(a2.x, a2.y); p1.y = pack_bf16x2(a2.z, a2.w);
            p1.z = pack_bf16x2(a3.x, a3.y); p1.w = pack_bf16x2(a3.z, a3.w);
            *(uint4*)&lds_a[tr * 16 + (tk >> 1) + 0] = p0;
            *(uint4*)&lds_a[tr * 16 + (tk >> 1) + 4] = p1;
        }
        __syncthreads();

        // ---- B fragment: lane<16 -> K k0..7,16..23, lane>=16 -> K k0+8..15,24..31 ----
        const int kb = (k0 + ((lane < 16) ? 0 : 8)) >> 3;  // uint4 index
        bfrag.q[0] = wrow[kb];
        bfrag.q[1] = wrow[kb + 2];

        // ---- 4 M-subtiles share the B fragment ----
#pragma unroll
        for (int s = 0; s < 4; ++s) {
            const int ab = s * 256 + abase0;
            afrag.q[0] = *(const uint4*)&lds_a[ab];
            afrag.q[1] = *(const uint4*)&lds_a[ab + 8];
            acc[s] = __builtin_amdgcn_wmma_f32_16x16x32_bf16(
                false, afrag.v, false, bfrag.v, (short)0, acc[s], false, false);
        }
        __syncthreads();
    }

    if (colok) {
        const float bb = BIAS ? bias[col] : 0.f;
        const int rbase = (lane < 16) ? 0 : 8;
#pragma unroll
        for (int s = 0; s < 4; ++s) {
            float* op = out + (size_t)(m_base + s * 16 + rbase) * LDO + col;
#pragma unroll
            for (int r = 0; r < 8; ++r)
                op[(size_t)r * LDO] = acc[s][r] + bb;
        }
    }
}

// ---------------------------------------------------------------------------
// Causal depthwise conv1d (k=4, left pad 3) + bias + SiLU.
// 4 channels per thread (b128 loads/stores), branchless edges.
// ---------------------------------------------------------------------------
__global__ void conv_silu_kernel(const float* __restrict__ xin, int ld_in,
                                 const float* __restrict__ w,
                                 const float* __restrict__ bias,
                                 float* __restrict__ out,
                                 int di, int L, int total4) {
    int idx = blockIdx.x * blockDim.x + threadIdx.x;
    if (idx >= total4) return;
    const int cpr = di >> 2;                   // channel-quads per row
    const int c4  = (idx % cpr) * 4;
    const int m   = idx / cpr;
    const int l   = m & (L - 1);               // L = 4096 (pow2)
    const int m0  = m - l;                     // b*L

    F4 wr[4];                                  // wr[ch].f[j] = w[c4+ch][j]
#pragma unroll
    for (int ch = 0; ch < 4; ++ch) wr[ch].v = ((const float4*)(w + c4 * 4))[ch];
    F4 acc; acc.v = *(const float4*)(bias + c4);

#pragma unroll
    for (int j = 0; j < 4; ++j) {
        const int lo  = l - 3 + j;
        const int loc = lo < 0 ? 0 : lo;       // clamped in-bounds address
        F4 xv; xv.v = *(const float4*)(xin + (size_t)(m0 + loc) * ld_in + c4);
        const bool ok = (lo >= 0);
#pragma unroll
        for (int ch = 0; ch < 4; ++ch)
            acc.f[ch] = fmaf(wr[ch].f[j], ok ? xv.f[ch] : 0.f, acc.f[ch]);
    }
    F4 o;
#pragma unroll
    for (int ch = 0; ch < 4; ++ch)
        o.f[ch] = acc.f[ch] / (1.f + __expf(-acc.f[ch]));   // silu
    *(float4*)(out + (size_t)m * di + c4) = o.v;
}

// ---------------------------------------------------------------------------
// delta = softplus(dt @ dt_w^T + dt_b); dt = first DTR cols of xdbl.
// 4 channels per thread; K=4/8 far too skinny for WMMA.
// ---------------------------------------------------------------------------
template <int DTR>
__global__ void delta_kernel(const float* __restrict__ xdbl, int ldx,
                             const float* __restrict__ dt_w,
                             const float* __restrict__ dt_b,
                             float* __restrict__ delta,
                             int di, int total4) {
    int idx = blockIdx.x * blockDim.x + threadIdx.x;
    if (idx >= total4) return;
    const int cpr = di >> 2;
    const int c4  = (idx % cpr) * 4;
    const int m   = idx / cpr;

    float xr[DTR];
#pragma unroll
    for (int r = 0; r < DTR; ++r) xr[r] = xdbl[(size_t)m * ldx + r];

    F4 acc; acc.v = *(const float4*)(dt_b + c4);
    const float* wbase = dt_w + c4 * DTR;      // 4 rows of DTR, contiguous
#pragma unroll
    for (int ch = 0; ch < 4; ++ch)
#pragma unroll
        for (int r = 0; r < DTR; ++r)
            acc.f[ch] = fmaf(xr[r], wbase[ch * DTR + r], acc.f[ch]);

    F4 o;
#pragma unroll
    for (int ch = 0; ch < 4; ++ch) {
        float v = acc.f[ch];
        o.f[ch] = (v > 20.f) ? v : log1pf(__expf(v));       // softplus
    }
    *(float4*)(delta + (size_t)m * di + c4) = o.v;
}

// ---------------------------------------------------------------------------
// SSM scan. One lane per (b, d, n): 16 lanes per (b,d) sequence, shuffle-
// reduce y = sum_n h*C each step. Fuses +xc*D and *silu(res) epilogue.
// ---------------------------------------------------------------------------
__global__ void ssm_scan_kernel(const float* __restrict__ delta,
                                const float* __restrict__ xc,
                                const float* __restrict__ xdbl, int ld_xdbl, int dtr,
                                const float* __restrict__ A_log,
                                const float* __restrict__ Dp,
                                const float* __restrict__ res, int ld_res,
                                float* __restrict__ y,
                                int di, int L) {
    const int t   = blockIdx.x * blockDim.x + threadIdx.x;
    const int n   = t & 15;
    const int seq = t >> 4;
    const int d   = seq % di;
    const int b   = seq / di;

    const float a  = -__expf(A_log[d * 16 + n]);
    const float Dd = Dp[d];
    float h = 0.f;
    const size_t mbase = (size_t)b * L;

    for (int l = 0; l < L; ++l) {
        const size_t m  = mbase + l;
        // prefetch next timestep's B/C row while this step computes
        const size_t mn = m + ((l + 1 < L) ? 1 : 0);
        __builtin_prefetch(&xdbl[mn * ld_xdbl], 0, 0);

        const float dl = delta[m * di + d];
        const float u  = xc[m * di + d];
        const float Bn = xdbl[m * ld_xdbl + dtr + n];
        const float Cn = xdbl[m * ld_xdbl + dtr + 16 + n];
        const float dA = __expf(dl * a);
        h = fmaf(dA, h, dl * u * Bn);
        float yp = h * Cn;
        yp += __shfl_xor(yp, 1, 16);
        yp += __shfl_xor(yp, 2, 16);
        yp += __shfl_xor(yp, 4, 16);
        yp += __shfl_xor(yp, 8, 16);
        if (n == 0) {
            float r = res[m * ld_res + d];
            float g = r / (1.f + __expf(-r));
            y[m * di + d] = (yp + u * Dd) * g;
        }
    }
}

// ---------------------------------------------------------------------------
// LayerNorm over 128 columns, one wave32 per row; optional residual add.
// ---------------------------------------------------------------------------
template <bool ADD>
__global__ __launch_bounds__(256)
void ln128_kernel(const float* __restrict__ in,
                  const float* __restrict__ g, const float* __restrict__ bvec,
                  const float* __restrict__ addv,
                  float* __restrict__ out, int M) {
    const int wave = blockIdx.x * (blockDim.x >> 5) + (threadIdx.x >> 5);
    const int lane = threadIdx.x & 31;
    if (wave >= M) return;
    const float* row = in + (size_t)wave * 128;
    float v[4], s = 0.f, q = 0.f;
#pragma unroll
    for (int i = 0; i < 4; ++i) { v[i] = row[lane + 32 * i]; s += v[i]; q += v[i] * v[i]; }
#pragma unroll
    for (int off = 16; off >= 1; off >>= 1) {
        s += __shfl_xor(s, off, 32);
        q += __shfl_xor(q, off, 32);
    }
    const float mean = s * (1.f / 128.f);
    const float var  = q * (1.f / 128.f) - mean * mean;
    const float rstd = rsqrtf(var + 1e-5f);
#pragma unroll
    for (int i = 0; i < 4; ++i) {
        int c = lane + 32 * i;
        float o = (v[i] - mean) * rstd * g[c] + bvec[c];
        if (ADD) o += addv[(size_t)wave * 128 + c];
        out[(size_t)wave * 128 + c] = o;
    }
}

// ---------------------------------------------------------------------------
// Host launcher
// ---------------------------------------------------------------------------
extern "C" void kernel_launch(void* const* d_in, const int* in_sizes, int n_in,
                              void* d_out, int out_size, void* d_ws, size_t ws_size,
                              hipStream_t stream) {
    (void)in_sizes; (void)n_in; (void)out_size; (void)ws_size;

    constexpr int Bsz = 4, L = 4096, M = Bsz * L;
    constexpr int OUT_C = 128;
    constexpr int DI1 = 128, DTR1 = 4, NX1 = 36;   // mamba1
    constexpr int DI2 = 256, DTR2 = 8, NX2 = 40;   // mamba2
    constexpr int LDX = 48;                        // padded x_dbl stride

    // Inputs (recursive insertion-order flatten of setup_inputs()):
    const float* x       = (const float*)d_in[0];
    const float* in_w1   = (const float*)d_in[1];   // [256,64]
    const float* conv_w1 = (const float*)d_in[2];   // [128,4]
    const float* conv_b1 = (const float*)d_in[3];
    const float* xp_w1   = (const float*)d_in[4];   // [36,128]
    const float* dt_w1   = (const float*)d_in[5];   // [128,4]
    const float* dt_b1   = (const float*)d_in[6];
    const float* A_log1  = (const float*)d_in[7];   // [128,16]
    const float* D1      = (const float*)d_in[8];
    const float* out_w1  = (const float*)d_in[9];   // [128,128]
    const float* in_w2   = (const float*)d_in[10];  // [512,128]
    const float* conv_w2 = (const float*)d_in[11];  // [256,4]
    const float* conv_b2 = (const float*)d_in[12];
    const float* xp_w2   = (const float*)d_in[13];  // [40,256]
    const float* dt_w2   = (const float*)d_in[14];  // [256,8]
    const float* dt_b2   = (const float*)d_in[15];
    const float* A_log2  = (const float*)d_in[16];  // [256,16]
    const float* D2      = (const float*)d_in[17];
    const float* out_w2  = (const float*)d_in[18];  // [128,256]
    const float* ln1_g   = (const float*)d_in[19];
    const float* ln1_b   = (const float*)d_in[20];
    const float* ln2_g   = (const float*)d_in[21];
    const float* ln2_b   = (const float*)d_in[22];
    const float* lin_w   = (const float*)d_in[23];  // [128,64]
    const float* lin_b   = (const float*)d_in[24];
    const float* lnr_g   = (const float*)d_in[25];
    const float* lnr_b   = (const float*)d_in[26];

    float* ws = (float*)d_ws;
    // fp32 scratch (floats/row): R 128 | A 512 | B 256 | C 48 | D 256 | E 256 | F 128
    float* BUF_R = ws;
    float* BUF_A = BUF_R + (size_t)M * 128;
    float* BUF_B = BUF_A + (size_t)M * 512;
    float* BUF_C = BUF_B + (size_t)M * 256;
    float* BUF_D = BUF_C + (size_t)M * LDX;
    float* BUF_E = BUF_D + (size_t)M * 256;
    float* BUF_F = BUF_E + (size_t)M * 256;

    // bf16 weight cache (ushorts) after fp32 scratch
    unsigned short* WB      = (unsigned short*)(BUF_F + (size_t)M * 128);
    unsigned short* lin_wb  = WB;                      // 128*64
    unsigned short* in_w1b  = lin_wb + 128 * 64;       // 256*64
    unsigned short* xp_w1b  = in_w1b + 256 * 64;       // 36*128
    unsigned short* out_w1b = xp_w1b + 36 * 128;       // 128*128
    unsigned short* in_w2b  = out_w1b + 128 * 128;     // 512*128
    unsigned short* xp_w2b  = in_w2b + 512 * 128;      // 40*256
    unsigned short* out_w2b = xp_w2b + 40 * 256;       // 128*256

    auto cvt = [&](const float* src, unsigned short* dst, int n) {
        f32_to_bf16_kernel<<<(n + 255) / 256, 256, 0, stream>>>(src, dst, n);
    };
    cvt(lin_w, lin_wb, 128 * 64);
    cvt(in_w1, in_w1b, 256 * 64);
    cvt(xp_w1, xp_w1b, 36 * 128);
    cvt(out_w1, out_w1b, 128 * 128);
    cvt(in_w2, in_w2b, 512 * 128);
    cvt(xp_w2, xp_w2b, 40 * 256);
    cvt(out_w2, out_w2b, 128 * 256);

    const dim3 blk(128);
    auto grid = [&](int N) { return dim3(M / 64, (N + 63) / 64); };

    // ---- residual linear branch: r = LN(x @ lin_w^T + lin_b) ----
    gemm_wmma_bf16<64, 128, true><<<grid(OUT_C), blk, 0, stream>>>(x, lin_wb, lin_b, BUF_R, OUT_C);
    ln128_kernel<false><<<M / 8, 256, 0, stream>>>(BUF_R, lnr_g, lnr_b, nullptr, BUF_R, M);

    // ---- mamba 1 (d_model=64, di=128) ----
    gemm_wmma_bf16<64, 256, false><<<grid(2 * DI1), blk, 0, stream>>>(x, in_w1b, nullptr, BUF_A, 2 * DI1);
    conv_silu_kernel<<<(M * DI1 / 4) / 256, 256, 0, stream>>>(BUF_A, 2 * DI1, conv_w1, conv_b1,
                                                              BUF_B, DI1, L, M * DI1 / 4);
    gemm_wmma_bf16<128, LDX, false><<<grid(NX1), blk, 0, stream>>>(BUF_B, xp_w1b, nullptr, BUF_C, NX1);
    delta_kernel<DTR1><<<(M * DI1 / 4) / 256, 256, 0, stream>>>(BUF_C, LDX, dt_w1, dt_b1,
                                                                BUF_D, DI1, M * DI1 / 4);
    ssm_scan_kernel<<<(Bsz * DI1 * 16) / 256, 256, 0, stream>>>(
        BUF_D, BUF_B, BUF_C, LDX, DTR1, A_log1, D1, BUF_A + DI1, 2 * DI1, BUF_E, DI1, L);
    gemm_wmma_bf16<128, 128, false><<<grid(OUT_C), blk, 0, stream>>>(BUF_E, out_w1b, nullptr, BUF_F, OUT_C);
    ln128_kernel<false><<<M / 8, 256, 0, stream>>>(BUF_F, ln1_g, ln1_b, nullptr, BUF_F, M);

    // ---- mamba 2 (d_model=128, di=256) ----
    gemm_wmma_bf16<128, 512, false><<<grid(2 * DI2), blk, 0, stream>>>(BUF_F, in_w2b, nullptr, BUF_A, 2 * DI2);
    conv_silu_kernel<<<(M * DI2 / 4) / 256, 256, 0, stream>>>(BUF_A, 2 * DI2, conv_w2, conv_b2,
                                                              BUF_B, DI2, L, M * DI2 / 4);
    gemm_wmma_bf16<256, LDX, false><<<grid(NX2), blk, 0, stream>>>(BUF_B, xp_w2b, nullptr, BUF_C, NX2);
    delta_kernel<DTR2><<<(M * DI2 / 4) / 256, 256, 0, stream>>>(BUF_C, LDX, dt_w2, dt_b2,
                                                                BUF_D, DI2, M * DI2 / 4);
    ssm_scan_kernel<<<(Bsz * DI2 * 16) / 256, 256, 0, stream>>>(
        BUF_D, BUF_B, BUF_C, LDX, DTR2, A_log2, D2, BUF_A + DI2, 2 * DI2, BUF_E, DI2, L);
    gemm_wmma_bf16<256, 128, false><<<grid(OUT_C), blk, 0, stream>>>(BUF_E, out_w2b, nullptr, BUF_F, OUT_C);

    // ---- final: out = r + LN2(h2) ----
    ln128_kernel<true><<<M / 8, 256, 0, stream>>>(BUF_F, ln2_g, ln2_b, BUF_R, (float*)d_out, M);
}
